// GINv2_7481833030168
// MI455X (gfx1250) — matrix operand order repaired and checked by reading, not compiled, not run on hardware
//
#include <hip/hip_runtime.h>

// GIN: 3x (scatter-add + fused 64x64x64 WMMA MLP) + pool + final WMMA linear.
// Target: MI455X gfx1250, wave32, v_wmma_f32_16x16x32_f16.
// Weights are staged into LDS pre-swizzled into the WMMA B-operand lane
// layout so each B tile is one contiguous 32B read per lane (2x ds_load_b128).

typedef __attribute__((ext_vector_type(16))) _Float16 v16h;
typedef __attribute__((ext_vector_type(8)))  float    v8f;

__device__ inline v8f wmma_f32_f16(v16h a, v16h b, v8f c) {
  // 8 args: (neg_a, A, neg_b, B, c_mod, C, reuse_a, reuse_b)
  return __builtin_amdgcn_wmma_f32_16x16x32_f16(false, a, false, b, (short)0, c,
                                                false, false);
}

// A operand (16x32 f16) from a row-major [16 x 64] f16 tile in LDS.
// Per ISA 16-bit A layout: lanes 0-15 -> M=lane, K in {kb..kb+7, kb+16..kb+23}
// with kb = 0 for lanes 0-15 and kb = 8 for lanes 16-31.
__device__ inline v16h load_a_tile(const _Float16* T, int lane, int kOff) {
  v16h a;
  const int r  = lane & 15;
  const int kb = (lane >> 4) * 8;
#pragma unroll
  for (int j = 0; j < 8; ++j) {
    a[j]     = T[r * 64 + kOff + kb + j];
    a[j + 8] = T[r * 64 + kOff + 16 + kb + j];
  }
  return a;
}

// Stage a (64 x NT*16) row-major f32 weight into LDS in B-operand swizzled
// layout: block (kt, nt) holds 32 lanes x 16 contiguous halves, where lane l
// element j = W[kt*32 + (l>=16?8:0) + (j&7) + (j>=8?16:0)][nt*16 + (l&15)].
template <int NT>
__device__ inline void stage_weights_swizzled(const float* __restrict__ W,
                                              _Float16* __restrict__ S,
                                              int tid) {
  const int total = 2 * NT * 32 * 16;  // == 64 * NT*16
  for (int idx = tid; idx < total; idx += 256) {
    const int j    = idx & 15;
    const int lane = (idx >> 4) & 31;
    const int blk  = idx >> 9;        // kt*NT + nt
    const int kt   = blk / NT;
    const int nt   = blk - kt * NT;
    const int k    = kt * 32 + ((lane >> 4) * 8) + (j & 7) + ((j >> 3) * 16);
    const int n    = nt * 16 + (lane & 15);
    S[idx] = (_Float16)W[k * (NT * 16) + n];
  }
}

template <int NT>
__device__ inline v16h load_b_sw(const _Float16* S, int lane, int kt, int nt) {
  return *(const v16h*)(S + (((kt * NT + nt) * 32) + lane) * 16);
}

__global__ __launch_bounds__(256) void gin_zero_kernel(float4* __restrict__ p,
                                                       int n4) {
  int i = blockIdx.x * blockDim.x + threadIdx.x;
  if (i < n4) p[i] = make_float4(0.f, 0.f, 0.f, 0.f);
}

// One thread per (edge, 4-channel group): gather h[src] (float4), atomic-add
// into agg[dst]. agg (12.8 MB) is L2-resident on a 192 MB L2.
__global__ __launch_bounds__(256) void gin_scatter_kernel(
    const float* __restrict__ h, const int* __restrict__ src,
    const int* __restrict__ dst, float* __restrict__ agg, int nE) {
  int gid = blockIdx.x * blockDim.x + threadIdx.x;
  int e = gid >> 4;
  if (e >= nE) return;
  int c = (gid & 15) * 4;
  int s = src[e];
  int d = dst[e];
  const float4 v = *(const float4*)(h + (size_t)s * 64 + c);
  float* out = agg + (size_t)d * 64 + c;
  atomicAdd(out + 0, v.x);
  atomicAdd(out + 1, v.y);
  atomicAdd(out + 2, v.z);
  atomicAdd(out + 3, v.w);
}

// Fused GIN MLP: h_out = relu( relu((h_in+agg) @ Wa + ba) @ Wb + bb )
// One wave handles a 16-row tile; 8 waves (256 threads) per block = 128 rows.
// h_out may alias agg: each wave reads its rows into LDS before writing them.
__global__ __launch_bounds__(256) void gin_mlp_kernel(
    const float* __restrict__ h_in, const float* __restrict__ agg,
    const float* __restrict__ wa, const float* __restrict__ ba,
    const float* __restrict__ wb, const float* __restrict__ bb,
    float* __restrict__ h_out, int n_rows) {
  __shared__ _Float16 s_wa[64 * 64];   // B-swizzled
  __shared__ _Float16 s_wb[64 * 64];   // B-swizzled
  __shared__ float    s_ba[64];
  __shared__ float    s_bb[64];
  __shared__ _Float16 s_t[8][16 * 64];  // per-wave activation tile

  const int tid = threadIdx.x;
  stage_weights_swizzled<4>(wa, s_wa, tid);
  stage_weights_swizzled<4>(wb, s_wb, tid);
  if (tid < 64) {
    s_ba[tid] = ba[tid];
    s_bb[tid] = bb[tid];
  }

  const int wave    = tid >> 5;
  const int lane    = tid & 31;
  const int rowBase = (blockIdx.x * 8 + wave) * 16;
  _Float16* T = s_t[wave];

  // Stage z = h_in + agg as f16 (coalesced), guarded for the row tail.
#pragma unroll
  for (int i = 0; i < 32; ++i) {
    int idx = i * 32 + lane;  // 0..1023 within 16x64 tile
    int r   = rowBase + (idx >> 6);
    int c   = idx & 63;
    float v = 0.f;
    if (r < n_rows) v = h_in[(size_t)r * 64 + c] + agg[(size_t)r * 64 + c];
    T[idx] = (_Float16)v;
  }
  __syncthreads();

  const int col = lane & 15;
  const int hi8 = (lane >> 4) * 8;

  // GEMM 1: t = relu(z @ Wa + ba)
  v16h a0 = load_a_tile(T, lane, 0);
  v16h a1 = load_a_tile(T, lane, 32);
  v8f acc[4];
#pragma unroll
  for (int nt = 0; nt < 4; ++nt) {
    v8f c = {};
    c = wmma_f32_f16(a0, load_b_sw<4>(s_wa, lane, 0, nt), c);
    c = wmma_f32_f16(a1, load_b_sw<4>(s_wa, lane, 1, nt), c);
    acc[nt] = c;
  }
  // Write relu(t+ba) back into T (C-layout -> row-major f16). Per-wave
  // region; DS ops are in-order within a wave.
#pragma unroll
  for (int nt = 0; nt < 4; ++nt) {
#pragma unroll
    for (int i = 0; i < 8; ++i) {
      float v = acc[nt][i] + s_ba[nt * 16 + col];
      v = fmaxf(v, 0.f);
      T[(i + hi8) * 64 + nt * 16 + col] = (_Float16)v;
    }
  }
  __syncthreads();

  // GEMM 2: out = relu(t @ Wb + bb)
  v16h d0 = load_a_tile(T, lane, 0);
  v16h d1 = load_a_tile(T, lane, 32);
#pragma unroll
  for (int nt = 0; nt < 4; ++nt) {
    v8f c = {};
    c = wmma_f32_f16(d0, load_b_sw<4>(s_wb, lane, 0, nt), c);
    c = wmma_f32_f16(d1, load_b_sw<4>(s_wb, lane, 1, nt), c);
#pragma unroll
    for (int i = 0; i < 8; ++i) {
      int r = rowBase + i + hi8;
      if (r < n_rows) {
        float v = c[i] + s_bb[nt * 16 + col];
        h_out[(size_t)r * 64 + nt * 16 + col] = fmaxf(v, 0.f);
      }
    }
  }
}

// pooled[batch[n]] += h[n]
__global__ __launch_bounds__(256) void gin_pool_kernel(
    const float* __restrict__ h, const int* __restrict__ batch,
    float* __restrict__ pooled, int n_rows) {
  int gid = blockIdx.x * blockDim.x + threadIdx.x;
  int n = gid >> 4;
  if (n >= n_rows) return;
  int c = (gid & 15) * 4;
  int g = batch[n];
  const float4 v = *(const float4*)(h + (size_t)n * 64 + c);
  float* out = pooled + (size_t)g * 64 + c;
  atomicAdd(out + 0, v.x);
  atomicAdd(out + 1, v.y);
  atomicAdd(out + 2, v.z);
  atomicAdd(out + 3, v.w);
}

// out(512x32) = pooled(512x64) @ lw(64x32) + lb. 32 row tiles -> 4 blocks.
__global__ __launch_bounds__(256) void gin_final_kernel(
    const float* __restrict__ pooled, const float* __restrict__ lw,
    const float* __restrict__ lb, float* __restrict__ out) {
  __shared__ _Float16 s_w[64 * 32];   // B-swizzled
  __shared__ float    s_b[32];
  __shared__ _Float16 s_t[8][16 * 64];

  const int tid = threadIdx.x;
  stage_weights_swizzled<2>(lw, s_w, tid);
  if (tid < 32) s_b[tid] = lb[tid];

  const int wave    = tid >> 5;
  const int lane    = tid & 31;
  const int rowBase = (blockIdx.x * 8 + wave) * 16;  // < 512 always
  _Float16* T = s_t[wave];
#pragma unroll
  for (int i = 0; i < 32; ++i) {
    int idx = i * 32 + lane;
    T[idx] = (_Float16)pooled[(size_t)(rowBase + (idx >> 6)) * 64 + (idx & 63)];
  }
  __syncthreads();

  const int col = lane & 15;
  const int hi8 = (lane >> 4) * 8;
  v16h a0 = load_a_tile(T, lane, 0);
  v16h a1 = load_a_tile(T, lane, 32);
#pragma unroll
  for (int nt = 0; nt < 2; ++nt) {
    v8f c = {};
    c = wmma_f32_f16(a0, load_b_sw<2>(s_w, lane, 0, nt), c);
    c = wmma_f32_f16(a1, load_b_sw<2>(s_w, lane, 1, nt), c);
#pragma unroll
    for (int i = 0; i < 8; ++i) {
      int r = rowBase + i + hi8;
      out[(size_t)r * 32 + nt * 16 + col] = c[i] + s_b[nt * 16 + col];
    }
  }
}

extern "C" void kernel_launch(void* const* d_in, const int* in_sizes, int n_in,
                              void* d_out, int out_size, void* d_ws,
                              size_t ws_size, hipStream_t stream) {
  (void)n_in; (void)out_size; (void)ws_size;
  const int N = in_sizes[0] / 64;   // 50000
  const int E = in_sizes[1] / 2;    // 800000
  const int NUM_GRAPHS = 512;

  const float* x     = (const float*)d_in[0];
  const int*   ei    = (const int*)d_in[1];
  const int*   batch = (const int*)d_in[2];
  const float* w0a = (const float*)d_in[3];
  const float* b0a = (const float*)d_in[4];
  const float* w0b = (const float*)d_in[5];
  const float* b0b = (const float*)d_in[6];
  const float* w1a = (const float*)d_in[7];
  const float* b1a = (const float*)d_in[8];
  const float* w1b = (const float*)d_in[9];
  const float* b1b = (const float*)d_in[10];
  const float* w2a = (const float*)d_in[11];
  const float* b2a = (const float*)d_in[12];
  const float* w2b = (const float*)d_in[13];
  const float* b2b = (const float*)d_in[14];
  const float* lw  = (const float*)d_in[15];
  const float* lb  = (const float*)d_in[16];

  const int* src = ei;
  const int* dst = ei + E;

  float* bufA   = (float*)d_ws;                  // N*64
  float* bufB   = bufA + (size_t)N * 64;         // N*64
  float* pooled = bufB + (size_t)N * 64;         // 512*64

  const int feat4   = N * 16;                    // float4 count of a buffer
  const int zblk    = (feat4 + 255) / 256;
  const int sblk    = (E * 16 + 255) / 256;
  const int mblk    = (N + 127) / 128;
  const int pblk    = (N * 16 + 255) / 256;
  const int pooled4 = NUM_GRAPHS * 16;

  // Layer 0: agg in bufA, h0 written over bufA
  gin_zero_kernel<<<zblk, 256, 0, stream>>>((float4*)bufA, feat4);
  gin_scatter_kernel<<<sblk, 256, 0, stream>>>(x, src, dst, bufA, E);
  gin_mlp_kernel<<<mblk, 256, 0, stream>>>(x, bufA, w0a, b0a, w0b, b0b, bufA, N);

  // Layer 1: agg in bufB, h1 written over bufB
  gin_zero_kernel<<<zblk, 256, 0, stream>>>((float4*)bufB, feat4);
  gin_scatter_kernel<<<sblk, 256, 0, stream>>>(bufA, src, dst, bufB, E);
  gin_mlp_kernel<<<mblk, 256, 0, stream>>>(bufA, bufB, w1a, b1a, w1b, b1b, bufB, N);

  // Layer 2: agg in bufA, h2 written over bufA
  gin_zero_kernel<<<zblk, 256, 0, stream>>>((float4*)bufA, feat4);
  gin_scatter_kernel<<<sblk, 256, 0, stream>>>(bufB, src, dst, bufA, E);
  gin_mlp_kernel<<<mblk, 256, 0, stream>>>(bufB, bufA, w2a, b2a, w2b, b2b, bufA, N);

  // Pool + final linear
  gin_zero_kernel<<<(pooled4 + 255) / 256, 256, 0, stream>>>((float4*)pooled,
                                                             pooled4);
  gin_pool_kernel<<<pblk, 256, 0, stream>>>(bufA, batch, pooled, N);
  gin_final_kernel<<<4, 256, 0, stream>>>(pooled, lw, lb, (float*)d_out);
}
